// SegBig_3642132267196
// MI455X (gfx1250) — matrix operand order repaired and checked by reading, not compiled, not run on hardware
//
#include <hip/hip_runtime.h>
#include <math.h>

// ---------------------------------------------------------------------------
// CDNA5 (gfx1250) implementation of the ConvPoint-style segmentation network.
// GEMMs go through v_wmma_f32_16x16x32_f16 (wave32 WMMA, f32 accumulate).
// ---------------------------------------------------------------------------

typedef __attribute__((ext_vector_type(16))) _Float16 v16h;
typedef __attribute__((ext_vector_type(8)))  float    v8f;

union AFrag { v16h v; _Float16 h[16]; };
union CFrag { v8f  v; float    f[8];  };
union HPack { uint4 u[2]; _Float16 h[16]; };

__device__ __forceinline__ int imin(int a, int b) { return a < b ? a : b; }

// ---------------------------------------------------------------------------
// KNN: one query point per thread; insertion-sorted top-K (K <= 16), ascending
// distance, ties keep the earlier index (matches jax.lax.top_k ordering).
// All point sets are prefixes of input_pts, batch stride = 8192*3.
// ---------------------------------------------------------------------------
__device__ __forceinline__ void knn_insert(float* bd, int* bi, int K, float d2, int j) {
    int p = K - 1;
    while (p > 0 && bd[p - 1] > d2) { bd[p] = bd[p - 1]; bi[p] = bi[p - 1]; --p; }
    bd[p] = d2; bi[p] = j;
}

__global__ __launch_bounds__(256)
void knn_kernel(const float* __restrict__ pts, int Nq, int Ns, int K,
                int* __restrict__ idx, int useTile)
{
    __shared__ float sp[256][3];
    const int t = blockIdx.x * 256 + threadIdx.x;
    const int total = 4 * Nq;
    const bool act = t < total;
    const int b = act ? (t / Nq) : 0;
    const int n = act ? (t % Nq) : 0;
    float qx = 0.f, qy = 0.f, qz = 0.f;
    if (act) {
        const float* q = pts + ((size_t)b * 8192 + n) * 3;
        qx = q[0]; qy = q[1]; qz = q[2];
    }
    float bd[16]; int bi[16];
    #pragma unroll
    for (int k = 0; k < 16; ++k) { bd[k] = 3.4e38f; bi[k] = 0; }
    float worst = 3.4e38f;

    if (useTile) {                    // Nq % 256 == 0 -> whole block shares b
        for (int j0 = 0; j0 < Ns; j0 += 256) {
            int j = j0 + threadIdx.x;
            if (j < Ns) {
                const float* s = pts + ((size_t)b * 8192 + j) * 3;
                sp[threadIdx.x][0] = s[0]; sp[threadIdx.x][1] = s[1]; sp[threadIdx.x][2] = s[2];
            }
            __syncthreads();
            int lim = imin(256, Ns - j0);
            for (int jj = 0; jj < lim; ++jj) {
                float dx = sp[jj][0] - qx, dy = sp[jj][1] - qy, dz = sp[jj][2] - qz;
                float d2 = dx * dx + dy * dy + dz * dz;
                if (d2 < worst) { knn_insert(bd, bi, K, d2, j0 + jj); worst = bd[K - 1]; }
            }
            __syncthreads();
        }
    } else if (act) {
        for (int j = 0; j < Ns; ++j) {
            const float* s = pts + ((size_t)b * 8192 + j) * 3;
            float dx = s[0] - qx, dy = s[1] - qy, dz = s[2] - qz;
            float d2 = dx * dx + dy * dy + dz * dz;
            if (d2 < worst) { knn_insert(bd, bi, K, d2, j); worst = bd[K - 1]; }
        }
    }
    if (act)
        for (int k = 0; k < K; ++k) idx[(size_t)t * K + k] = bi[k];
}

// ---------------------------------------------------------------------------
// Per-point kernel: gather rel. positions, normalize, run the 48->32->16->16
// MLP per neighbor, stage neighbor features in LDS, and emit the f16 matrix
// f[row, c*16+m] = (1/K) * sum_k nb_f[k,c] * mlp[k,m], zero-padded to Ktpad.
// Rows past Mtot (grid round-up) are written as zeros so the GEMM A-tile
// loads need no guards at all. One block (128 threads) per row.
// ---------------------------------------------------------------------------
__global__ __launch_bounds__(128)
void feat_kernel(_Float16* __restrict__ fint,
                 const float* __restrict__ feats, int fstride, int foff, int cin,
                 const float* __restrict__ pts, int Nq, int Ns, int K,
                 const int* __restrict__ idx,
                 const float* __restrict__ centers,
                 const float* __restrict__ l1w, const float* __restrict__ l1b,
                 const float* __restrict__ l2w, const float* __restrict__ l2b,
                 const float* __restrict__ l3w, const float* __restrict__ l3b,
                 int rowStart, int Mtot, int Ktpad)
{
    __shared__ float rel[16][3];
    __shared__ float r2s[16];
    __shared__ float invs;
    __shared__ float dl[16][16];
    __shared__ float nbf[1024];

    const int tid = threadIdx.x;
    const int row = rowStart + blockIdx.x;
    const int Kt = cin * 16;
    _Float16* out = fint + (size_t)blockIdx.x * Ktpad;

    if (row >= Mtot) {                 // zero-fill padded rows (uniform branch)
        for (int i = tid; i < Ktpad; i += 128) out[i] = (_Float16)0.f;
        return;
    }
    const int b = row / Nq;
    const int n = row % Nq;

    const float* q = pts + ((size_t)b * 8192 + n) * 3;
    const float qx = q[0], qy = q[1], qz = q[2];

    if (tid < K) {
        int j = idx[(size_t)row * K + tid];
        const float* s = pts + ((size_t)b * 8192 + j) * 3;
        float dx = s[0] - qx, dy = s[1] - qy, dz = s[2] - qz;
        rel[tid][0] = dx; rel[tid][1] = dy; rel[tid][2] = dz;
        r2s[tid] = dx * dx + dy * dy + dz * dz;
    }
    __syncthreads();
    if (tid == 0) {
        float m = 0.f;
        for (int k = 0; k < K; ++k) m = fmaxf(m, r2s[k]);
        float mx = sqrtf(m);
        invs = (mx == 0.f) ? 1.f : 1.f / mx;
    }
    __syncthreads();
    if (tid < K) {
        float inv = invs;
        float in[48];
        #pragma unroll
        for (int d0 = 0; d0 < 3; ++d0) {
            float rv = rel[tid][d0] * inv;
            #pragma unroll
            for (int c = 0; c < 16; ++c) in[d0 * 16 + c] = rv - centers[d0 * 16 + c];
        }
        float h1[32];
        for (int j = 0; j < 32; ++j) {
            float s = l1b[j];
            #pragma unroll 8
            for (int i = 0; i < 48; ++i) s += in[i] * l1w[i * 32 + j];
            h1[j] = fmaxf(s, 0.f);
        }
        float h2[16];
        for (int j = 0; j < 16; ++j) {
            float s = l2b[j];
            #pragma unroll 8
            for (int i = 0; i < 32; ++i) s += h1[i] * l2w[i * 16 + j];
            h2[j] = fmaxf(s, 0.f);
        }
        const float invK = 1.f / (float)K;
        for (int j = 0; j < 16; ++j) {
            float s = l3b[j];
            #pragma unroll 8
            for (int i = 0; i < 16; ++i) s += h2[i] * l3w[i * 16 + j];
            dl[tid][j] = fmaxf(s, 0.f) * invK;
        }
    }
    // stage neighbor features: nbf[k*cin + c]
    for (int i = tid; i < K * cin; i += 128) {
        int k = i / cin;
        int c = i - k * cin;
        int j = idx[(size_t)row * K + k];
        nbf[i] = feats[((size_t)b * Ns + j) * fstride + foff + c];
    }
    __syncthreads();

    for (int i = tid; i < Kt; i += 128) {
        int c = i >> 4, m = i & 15;
        float s = 0.f;
        for (int k = 0; k < K; ++k) s += nbf[k * cin + c] * dl[k][m];
        out[i] = (_Float16)s;
    }
    for (int i = Kt + tid; i < Ktpad; i += 128)   // K padding (e.g. 48 -> 64)
        out[i] = (_Float16)0.f;
}

// ---------------------------------------------------------------------------
// WMMA GEMM: C[M, Cout] = A_f16[M, Ktpad] * W_f32[Ktot, Cout] (f32 accum).
// 64x64 tile per workgroup, 4 waves, each wave: 16 rows x 64 cols
// (4 x v_wmma_f32_16x16x32_f16 per BK=32 step). A is pre-padded with zeros
// (rows and K), so its tile loads are unguarded vector b128 ops.
// Fragment layouts per CDNA5 ISA 7.12.2.
// ---------------------------------------------------------------------------
__global__ __launch_bounds__(128)
void wmma_gemm_kernel(const _Float16* __restrict__ A, const float* __restrict__ W,
                      float* __restrict__ C, int rowStart,
                      int Mtot, int Ktot, int Ktpad, int Cout)
{
    __shared__ _Float16 As[64][40];   // 64 rows x BK=32 (+8 pad), 80B rows
    __shared__ _Float16 Bs[32][72];   // BK=32 rows x 64 cols (+8 pad), 144B rows

    const int tid  = threadIdx.x;
    const int wave = tid >> 5;
    const int lane = tid & 31;
    const int l    = lane & 15;
    const int hi   = lane >> 4;
    const int rowBlk = blockIdx.x * 64;
    const int colBlk = blockIdx.y * 64;

    // A-tile loader coords: 64 rows x 32 halves, 16 halves (32B) per thread
    const int ar  = tid >> 1;
    const int acb = (tid & 1) * 16;
    const _Float16* Abase = A + (size_t)(rowBlk + ar) * Ktpad + acb;
    // B-tile loader coords: 32 rows x 64 cols, 16 floats->halves per thread
    const int bkr = tid >> 2;
    const int bcb = (tid & 3) * 16;

    CFrag acc[4];
    #pragma unroll
    for (int i = 0; i < 4; ++i)
        #pragma unroll
        for (int j = 0; j < 8; ++j) acc[i].f[j] = 0.f;

    for (int kk = 0; kk < Ktpad; kk += 32) {
        {   // A tile: unguarded vector loads (buffer is zero-padded)
            const uint4* Ap = (const uint4*)(Abase + kk);
            uint4 q0 = Ap[0];
            uint4 q1 = Ap[1];
            if (kk + 32 < Ktpad)        // prefetch next k-tile (global_prefetch)
                __builtin_prefetch((const void*)(Abase + kk + 32), 0, 1);
            *((uint4*)&As[ar][acb])     = q0;
            *((uint4*)&As[ar][acb + 8]) = q1;
        }
        {   // B tile: one uniform guard per thread, f32 -> f16 convert
            HPack p;
            if (kk + bkr < Ktot) {
                const float4* Wp = (const float4*)(W + (size_t)(kk + bkr) * Cout + colBlk + bcb);
                #pragma unroll
                for (int v = 0; v < 4; ++v) {
                    float4 f = Wp[v];
                    p.h[v * 4 + 0] = (_Float16)f.x;
                    p.h[v * 4 + 1] = (_Float16)f.y;
                    p.h[v * 4 + 2] = (_Float16)f.z;
                    p.h[v * 4 + 3] = (_Float16)f.w;
                }
            } else {
                p.u[0] = make_uint4(0, 0, 0, 0);
                p.u[1] = make_uint4(0, 0, 0, 0);
            }
            *((uint4*)&Bs[bkr][bcb])     = p.u[0];
            *((uint4*)&Bs[bkr][bcb + 8]) = p.u[1];
        }
        __syncthreads();

        // A fragment: lane l holds M = l; half e -> K = 16*(e>>3) + 8*hi + (e&7)
        AFrag a;
        #pragma unroll
        for (int e = 0; e < 16; ++e) {
            int ke = ((e >> 3) << 4) + (hi << 3) + (e & 7);
            a.h[e] = As[wave * 16 + l][ke];
        }
        #pragma unroll
        for (int nt = 0; nt < 4; ++nt) {
            // B fragment: lane l holds N = l; half e -> K = 16*(e>>3) + 8*hi + (e&7)
            AFrag bf;
            #pragma unroll
            for (int e = 0; e < 16; ++e) bf.h[e] = Bs[((e >> 3) << 4) + (hi << 3) + (e & 7)][nt * 16 + l];
            acc[nt].v = __builtin_amdgcn_wmma_f32_16x16x32_f16(
                false, a.v, false, bf.v, (short)0, acc[nt].v, false, false);
        }
        __syncthreads();
    }
    // C/D layout: VGPR r, lanes 0-15 -> M=r, lanes 16-31 -> M=r+8; N = lane&15
    #pragma unroll
    for (int nt = 0; nt < 4; ++nt) {
        #pragma unroll
        for (int r = 0; r < 8; ++r) {
            int mRow = wave * 16 + r + 8 * hi;
            int grow = rowStart + rowBlk + mRow;
            int col  = colBlk + nt * 16 + l;
            if (grow < Mtot)
                C[(size_t)grow * Cout + col] = acc[nt].f[r];
        }
    }
}

// ---------------------------------------------------------------------------
// BatchNorm over all M rows (axis (0,1) in the reference), then ReLU write
// into the (possibly concat) destination with stride/offset.
// ---------------------------------------------------------------------------
__global__ __launch_bounds__(256)
void bn_stats_kernel(const float* __restrict__ x, int M, int C, float* __restrict__ stats)
{
    __shared__ float ss[256], ss2[256];
    const int c = blockIdx.x;
    float s = 0.f, s2 = 0.f;
    for (int r = threadIdx.x; r < M; r += 256) {
        float v = x[(size_t)r * C + c];
        s += v; s2 += v * v;
    }
    ss[threadIdx.x] = s; ss2[threadIdx.x] = s2;
    __syncthreads();
    for (int st = 128; st > 0; st >>= 1) {
        if (threadIdx.x < st) { ss[threadIdx.x] += ss[threadIdx.x + st]; ss2[threadIdx.x] += ss2[threadIdx.x + st]; }
        __syncthreads();
    }
    if (threadIdx.x == 0) {
        float m = ss[0] / (float)M;
        stats[2 * c]     = m;
        stats[2 * c + 1] = ss2[0] / (float)M - m * m;
    }
}

__global__ __launch_bounds__(256)
void bn_apply_kernel(const float* __restrict__ x, int M, int C,
                     const float* __restrict__ stats,
                     const float* __restrict__ g, const float* __restrict__ bb,
                     float* __restrict__ dst, int dstride, int doff)
{
    int t = blockIdx.x * 256 + threadIdx.x;
    if (t >= M * C) return;
    int r = t / C, c = t - r * C;
    float m = stats[2 * c], v = stats[2 * c + 1];
    float y = (x[t] - m) * rsqrtf(v + 1e-5f) * g[c] + bb[c];
    dst[(size_t)r * dstride + doff + c] = fmaxf(y, 0.f);
}

// Final FC: out[M,13] = x[M,128] @ fcw[128,13] + fcb
__global__ __launch_bounds__(256)
void fc_kernel(const float* __restrict__ x, const float* __restrict__ w,
               const float* __restrict__ bias, float* __restrict__ out, int M)
{
    int t = blockIdx.x * 256 + threadIdx.x;
    if (t >= M * 13) return;
    int r = t / 13, o = t - r * 13;
    float s = bias[o];
    const float* xr = x + (size_t)r * 128;
    #pragma unroll 8
    for (int c = 0; c < 128; ++c) s += xr[c] * w[c * 13 + o];
    out[t] = s;
}

// ---------------------------------------------------------------------------
// Host orchestration
// ---------------------------------------------------------------------------
extern "C" void kernel_launch(void* const* d_in, const int* in_sizes, int n_in,
                              void* d_out, int out_size, void* d_ws, size_t ws_size,
                              hipStream_t stream)
{
    (void)in_sizes; (void)n_in; (void)out_size; (void)ws_size;
    const float* x_in = (const float*)d_in[0];    // (4,8192,3)
    const float* pts  = (const float*)d_in[1];    // (4,8192,3)

    // ---- workspace carve (all offsets deterministic) ----
    char* ws = (char*)d_ws;
    size_t off = 0;
    auto carve = [&](size_t bytes) -> char* {
        char* p = ws + off;
        off += (bytes + 255) & ~(size_t)255;
        return p;
    };
    float* bufX0D = (float*)carve((size_t)4 * 8192 * 128 * 4);
    float* bufX1D = (float*)carve((size_t)4 * 2048 * 128 * 4);
    float* bufX2D = (float*)carve((size_t)4 * 1024 * 128 * 4);
    float* bufX3D = (float*)carve((size_t)4 * 256  * 128 * 4);
    float* bufX4D = (float*)carve((size_t)4 * 64   * 256 * 4);
    float* bufX5D = (float*)carve((size_t)4 * 16   * 256 * 4);
    float* bufX6  = (float*)carve((size_t)4 * 8    * 128 * 4);
    int*   idxBuf = (int*)  carve((size_t)4 * 8192 * 16 * 4);
    const size_t FINT_HALVES = (size_t)16 * 1024 * 1024;          // 32 MB chunk
    _Float16* fint = (_Float16*)carve(FINT_HALVES * 2);
    float* preBN  = (float*)carve((size_t)32768 * 64 * 4);
    float* stats  = (float*)carve((size_t)2 * 128 * 4);

    float* bufs[7]         = {bufX0D, bufX1D, bufX2D, bufX3D, bufX4D, bufX5D, bufX6};
    const int bufStride[7] = {128, 128, 128, 128, 256, 256, 128};

    // {cin, cout, K, Nq, Ns, srcBuf(-1=input x), srcOff, dstBuf, dstOff}
    static const int LT[13][9] = {
        {  3,  64, 16, 8192, 8192, -1,   0, 0,  64},   // cv0
        { 64,  64, 16, 2048, 8192,  0,  64, 1,  64},   // cv1
        { 64,  64, 16, 1024, 2048,  1,  64, 2,  64},   // cv2
        { 64,  64, 16,  256, 1024,  2,  64, 3,  64},   // cv3
        { 64, 128,  8,   64,  256,  3,  64, 4, 128},   // cv4
        {128, 128,  8,   16,   64,  4, 128, 5, 128},   // cv5
        {128, 128,  4,    8,   16,  5, 128, 6,   0},   // cv6
        {128, 128,  4,   16,    8,  6,   0, 5,   0},   // cv5d
        {256, 128,  4,   64,   16,  5,   0, 4,   0},   // cv4d
        {256,  64,  4,  256,   64,  4,   0, 3,   0},   // cv3d
        {128,  64,  8, 1024,  256,  3,   0, 2,   0},   // cv2d
        {128,  64,  8, 2048, 1024,  2,   0, 1,   0},   // cv1d
        {128,  64,  8, 8192, 2048,  1,   0, 0,   0},   // cv0d
    };

    for (int li = 0; li < 13; ++li) {
        const int cin = LT[li][0], cout = LT[li][1], K = LT[li][2];
        const int Nq = LT[li][3], Ns = LT[li][4];
        const int sb = LT[li][5], soffc = LT[li][6];
        const int db = LT[li][7], doffc = LT[li][8];

        const float* src; int sstride;
        if (sb < 0) { src = x_in; sstride = 3; }
        else        { src = bufs[sb]; sstride = bufStride[sb]; }
        float* dst = bufs[db];
        const int dstride = bufStride[db];

        const int M     = 4 * Nq;
        const int Kt    = cin * 16;
        const int Ktpad = (Kt + 31) & ~31;
        const int pb = 2 + li * 10;     // param base in d_in (insertion order)
        const float* w       = (const float*)d_in[pb + 0];
        const float* centers = (const float*)d_in[pb + 1];
        const float* l1w     = (const float*)d_in[pb + 2];
        const float* l1b     = (const float*)d_in[pb + 3];
        const float* l2w     = (const float*)d_in[pb + 4];
        const float* l2b     = (const float*)d_in[pb + 5];
        const float* l3w     = (const float*)d_in[pb + 6];
        const float* l3b     = (const float*)d_in[pb + 7];
        const float* bng     = (const float*)d_in[pb + 8];
        const float* bnb     = (const float*)d_in[pb + 9];

        // KNN
        const int useTile = (Nq % 256 == 0) ? 1 : 0;
        knn_kernel<<<dim3((4 * Nq + 255) / 256), dim3(256), 0, stream>>>(
            pts, Nq, Ns, K, idxBuf, useTile);

        // chunk rows so the padded f16 intermediate fits the 32MB scratch
        int chRows = (int)(FINT_HALVES / (size_t)Ktpad);
        chRows &= ~63;
        if (chRows > M) chRows = M;
        if (chRows < 64) chRows = 64;

        for (int rs = 0; rs < M; rs += chRows) {
            int rows = (M - rs < chRows) ? (M - rs) : chRows;
            int rowsPad = (rows + 63) & ~63;
            feat_kernel<<<dim3(rowsPad), dim3(128), 0, stream>>>(
                fint, src, sstride, soffc, cin, pts, Nq, Ns, K, idxBuf,
                centers, l1w, l1b, l2w, l2b, l3w, l3b, rs, M, Ktpad);
            dim3 g2(rowsPad / 64, cout / 64);
            wmma_gemm_kernel<<<g2, dim3(128), 0, stream>>>(
                fint, w, preBN, rs, M, Kt, Ktpad, cout);
        }

        bn_stats_kernel<<<dim3(cout), dim3(256), 0, stream>>>(preBN, M, cout, stats);
        const int tot = M * cout;
        bn_apply_kernel<<<dim3((tot + 255) / 256), dim3(256), 0, stream>>>(
            preBN, M, cout, stats, bng, bnb, dst, dstride, doffc);
    }

    // Final FC: x0d (32768 x 128) @ fcw (128 x 13) + fcb
    const float* fcw = (const float*)d_in[2 + 13 * 10 + 0];
    const float* fcb = (const float*)d_in[2 + 13 * 10 + 1];
    const int M0 = 4 * 8192;
    fc_kernel<<<dim3((M0 * 13 + 255) / 256), dim3(256), 0, stream>>>(
        bufX0D, fcw, fcb, (float*)d_out, M0);
}
